// _Readout_36747740184856
// MI455X (gfx1250) — matrix operand order repaired and checked by reading, not compiled, not run on hardware
//
#include <hip/hip_runtime.h>

typedef __attribute__((ext_vector_type(16))) _Float16 v16h;
typedef __attribute__((ext_vector_type(8)))  _Float16 v8h;
typedef __attribute__((ext_vector_type(4)))  _Float16 v4h;
typedef __attribute__((ext_vector_type(8)))  float    v8f;
typedef __attribute__((ext_vector_type(4)))  float    v4f;

#define DD      128              // D
#define MT      4                // 16-row M-tiles per block iteration
#define ROWS    (MT * 16)        // 64 rows per iteration
#define APITCH  392              // f16 pitch of LDS A-tile row (stride % 64 dwords = 4)
#define RP32    384              // f32 pitch of raw staging row
#define RAWB    (ROWS * RP32 * 4)        // 98304 B per raw buffer
#define ATILEB  (ROWS * APITCH * 2)      // 50176 B
#define PARTB   (8 * ROWS * 4)           // 2048 B
#define SMEMB   (2 * RAWB + ATILEB + PARTB)   // 248832 B (< 320 KB WGP LDS)

__device__ __forceinline__ float gelu_exact(float x) {
    return 0.5f * x * (1.0f + erff(x * 0.7071067811865475f));
}

// Issue one tile's worth of async global->LDS copies (raw f32, 16B per lane per op).
// AS3->generic lowering is {aperture_hi32, lds_offset}, so low 32 bits of the
// generic pointer are the LDS address the async op needs in its VDST VGPR.
__device__ __forceinline__ void stage_async(
    const float* __restrict__ nodes, const float* __restrict__ z_local,
    const float* __restrict__ z_meta, const int* __restrict__ batch,
    long long rowBase, float* rawbuf, int tid)
{
    const unsigned ldsBase = (unsigned)(unsigned long long)(void*)rawbuf;
    for (int q = tid; q < ROWS * 96; q += 256) {   // 96 float4 per row
        const int row = q / 96;
        const int c4  = q - row * 96;
        const int j0  = c4 * 4;                    // 0..380
        const int seg = j0 >> 7;                   // 0 nodes, 1 z_local, 2 z_meta
        const int off = j0 & 127;
        const long long grow = rowBase + row;
        const float* src =
            (seg == 0) ? (nodes + grow * DD + off) :
            (seg == 1) ? (z_local + (long long)batch[grow] * DD + off) :
                         (z_meta + off);
        const unsigned dst = ldsBase + (unsigned)(row * RP32 + j0) * 4u;
        asm volatile("global_load_async_to_lds_b128 %0, %1, off"
                     :: "v"(dst), "v"(src) : "memory");
    }
}

extern "C" __global__ void __launch_bounds__(256)
readout_wmma(const float* __restrict__ nodes,
             const float* __restrict__ z_local,
             const float* __restrict__ z_meta,
             const float* __restrict__ W1,
             const float* __restrict__ b1,
             const float* __restrict__ W2,
             const float* __restrict__ b2,
             const int*   __restrict__ batch,
             float*       __restrict__ out,
             int n_iter)
{
    extern __shared__ __align__(16) char smem[];
    float*    raw0  = (float*)smem;                      // [64][384] f32
    float*    raw1  = (float*)(smem + RAWB);             // [64][384] f32
    _Float16* atile = (_Float16*)(smem + 2 * RAWB);      // [64][APITCH] f16
    float*    part  = (float*)(smem + 2 * RAWB + ATILEB);

    const int tid  = threadIdx.x;
    const int lane = tid & 31;
    const int wave = tid >> 5;        // this wave's N-tile (0..7)
    const int g    = lane >> 4;
    const int l    = lane & 15;

    const int   col   = wave * 16 + l;
    const float bb    = b1[col];
    const float ww    = W2[col];
    const float bias2 = b2[0];

    // ---- This wave's 12 B fragments, built once, kept resident (96 VGPRs). ----
    v16h breg[12];
    #pragma unroll
    for (int kc = 0; kc < 12; ++kc) {
        const int kbase = kc * 32 + g * 8;
        #pragma unroll
        for (int j = 0; j < 16; ++j)
            breg[kc][j] = (_Float16)W1[(kbase + (j < 8 ? j : j + 8)) * DD + col];
    }

    // ---- Prologue: async-prefetch first tile into raw0 ----
    const int it0 = blockIdx.x;
    if (it0 < n_iter)
        stage_async(nodes, z_local, z_meta, batch, (long long)it0 * ROWS, raw0, tid);

    int p = 0;
    for (int it = it0; it < n_iter; it += gridDim.x) {
        float* rawc = p ? raw1 : raw0;
        float* rawn = p ? raw0 : raw1;
        const long long rowBase = (long long)it * ROWS;

        // Current raw buffer's async copies complete (all waves), then prefetch next.
        asm volatile("s_wait_asynccnt 0x0" ::: "memory");
        __syncthreads();
        const int itn = it + gridDim.x;
        if (itn < n_iter)
            stage_async(nodes, z_local, z_meta, batch, (long long)itn * ROWS, rawn, tid);

        // ---- Convert raw f32 -> f16 A-tile (LDS -> LDS, overlapped with prefetch) ----
        for (int q = tid; q < ROWS * 96; q += 256) {
            const int row = q / 96;
            const int j0  = (q - row * 96) * 4;
            v4f v = *(const v4f*)(rawc + row * RP32 + j0);
            v4h h;
            h[0] = (_Float16)v[0]; h[1] = (_Float16)v[1];
            h[2] = (_Float16)v[2]; h[3] = (_Float16)v[3];
            *(v4h*)(atile + row * APITCH + j0) = h;
        }
        __syncthreads();

        // ---- 48 WMMAs: 12 K-chunks x 4 M-tiles, f32 accumulate ----
        v8f acc[MT];
        #pragma unroll
        for (int mt = 0; mt < MT; ++mt) acc[mt] = (v8f)0.0f;

        #pragma unroll
        for (int kc = 0; kc < 12; ++kc) {
            const int k0 = kc * 32;
            #pragma unroll
            for (int mt = 0; mt < MT; ++mt) {
                const _Float16* ap = atile + (mt * 16 + l) * APITCH + k0 + g * 8;
                v8h alo = *(const v8h*)ap;
                v8h ahi = *(const v8h*)(ap + 16);
                v16h afrag = __builtin_shufflevector(alo, ahi,
                    0,1,2,3,4,5,6,7,8,9,10,11,12,13,14,15);
                acc[mt] = __builtin_amdgcn_wmma_f32_16x16x32_f16(
                    /*neg_a=*/false, afrag, /*neg_b=*/false, breg[kc],
                    /*c_mod=*/(short)0, acc[mt],
                    /*reuse_a=*/false, /*reuse_b=*/false);
            }
        }

        // ---- Epilogue: gelu(acc + b1) . W2 on this wave's 16 columns ----
        #pragma unroll
        for (int mt = 0; mt < MT; ++mt) {
            float s[8];
            #pragma unroll
            for (int r = 0; r < 8; ++r)
                s[r] = gelu_exact(acc[mt][r] + bb) * ww;
            #pragma unroll
            for (int r = 0; r < 8; ++r) {
                float v = s[r];
                v += __shfl_xor(v, 1, 32);
                v += __shfl_xor(v, 2, 32);
                v += __shfl_xor(v, 4, 32);
                v += __shfl_xor(v, 8, 32);
                if (l == r) part[wave * ROWS + mt * 16 + g * 8 + r] = v;
            }
        }
        __syncthreads();

        if (tid < ROWS) {
            float v = bias2;
            #pragma unroll
            for (int w = 0; w < 8; ++w) v += part[w * ROWS + tid];
            out[rowBase + tid] = v;
        }
        __syncthreads();   // protect atile/part before next iteration

        p ^= 1;
    }
}

extern "C" void kernel_launch(void* const* d_in, const int* in_sizes, int n_in,
                              void* d_out, int out_size, void* d_ws, size_t ws_size,
                              hipStream_t stream) {
    const float* nodes  = (const float*)d_in[0];
    const float* zl     = (const float*)d_in[1];
    const float* zm     = (const float*)d_in[2];
    const float* W1     = (const float*)d_in[3];
    const float* b1     = (const float*)d_in[4];
    const float* W2     = (const float*)d_in[5];
    const float* b2     = (const float*)d_in[6];
    const int*   batch  = (const int*)d_in[7];
    float* out = (float*)d_out;

    const int rows   = in_sizes[0] / DD;    // C = 1,000,000
    const int n_iter = rows / ROWS;         // 15625, exact for this problem

    (void)hipFuncSetAttribute((const void*)readout_wmma,
                              hipFuncAttributeMaxDynamicSharedMemorySize,
                              (int)SMEMB);

    int blocks = n_iter < 1024 ? n_iter : 1024;
    readout_wmma<<<blocks, 256, SMEMB, stream>>>(
        nodes, zl, zm, W1, b1, W2, b2, batch, out, n_iter);
}
